// SelfAttention_65627100283196
// MI455X (gfx1250) — compile-verified, matrix operand order checked by reading, not run on hardware
//
#include <hip/hip_runtime.h>
#include <cstdint>

#define S_   2048
#define DIM_ 1536
#define NH_  12
#define HD_  128
#define POOL_ 128
#define QT   64

typedef __attribute__((ext_vector_type(16))) __bf16 v16bf;
typedef __attribute__((ext_vector_type(8)))  float  v8f;

union FragB { v16bf v; uint4 q[2]; unsigned short s[16]; };
union FragF { v8f   v; float f[8]; };

__device__ __forceinline__ unsigned short f32_to_bf16(float f) {
    unsigned int u = __float_as_uint(f);
    unsigned int r = (u + 0x7FFFu + ((u >> 16) & 1u)) >> 16;
    return (unsigned short)r;
}

__device__ __forceinline__ v8f wmma_bf16(const FragB& a, const FragB& b, const FragF& c) {
    return __builtin_amdgcn_wmma_f32_16x16x32_bf16(false, a.v, false, b.v,
                                                   (short)0, c.v, false, false);
}

// async global -> LDS, 16B per lane, tracked by ASYNCcnt
__device__ __forceinline__ void async_b128(unsigned lds_addr, uint64_t gaddr) {
    asm volatile("global_load_async_to_lds_b128 %0, %1, off"
                 :: "v"(lds_addr), "v"(gaddr) : "memory");
}
__device__ __forceinline__ void wait_async0() {
    asm volatile("s_wait_asynccnt 0" ::: "memory");
}
__device__ __forceinline__ unsigned lds_off(const void* p) {
    return (unsigned)(size_t)p;   // LDS generic address: low 32 bits = LDS byte offset
}

// ---------------------------------------------------------------- conversion
__global__ __launch_bounds__(256) void cvt_bf16_kernel(const float* __restrict__ in,
                                                       unsigned short* __restrict__ out, int n) {
    int i = blockIdx.x * 256 + threadIdx.x;
    if (i < n) out[i] = f32_to_bf16(in[i]);
}

// ---------------------------------------------------------------- transpose + convert
// in: f32 [R][C] row-major -> out: bf16 [C][R]
__global__ __launch_bounds__(256) void transpose_cvt_kernel(const float* __restrict__ in,
                                                            unsigned short* __restrict__ out,
                                                            int R, int C) {
    __shared__ unsigned short tile[32][33];
    int c0 = blockIdx.x * 32, r0 = blockIdx.y * 32;
    int tx = threadIdx.x & 31, ty = threadIdx.x >> 5;
    for (int rr = ty; rr < 32; rr += 8)
        tile[rr][tx] = f32_to_bf16(in[(size_t)(r0 + rr) * C + c0 + tx]);
    __syncthreads();
    for (int rr = ty; rr < 32; rr += 8)
        out[(size_t)(c0 + rr) * R + r0 + tx] = tile[tx][rr];
}

// ---------------------------------------------------------------- WMMA GEMM (async double-buffered)
// C[M,N] (f32) = A[M,K] (bf16 row-major) x Bt[N,K] (bf16: B pre-transposed)
// block tile 128x128, 8 waves (2 along M x 4 along N), K-step 32
__global__ __launch_bounds__(256) void gemm_bf16_wmma(const unsigned short* __restrict__ A,
                                                      const unsigned short* __restrict__ Bt,
                                                      float* __restrict__ C,
                                                      int M, int N, int K) {
    __shared__ __align__(16) unsigned short Abuf[2][128][32];   // [m][k]
    __shared__ __align__(16) unsigned short Bbuf[2][128][32];   // [n][k]

    const int tid  = threadIdx.x;
    const int lane = tid & 31;
    const int wave = tid >> 5;
    const int wm   = wave >> 2;        // 0..1 -> 64 rows each
    const int wn   = wave & 3;         // 0..3 -> 32 cols each
    const int lrow = lane & 15;
    const int khalf = (lane < 16) ? 0 : 8;
    const int loff  = (lane < 16) ? 0 : 8;
    const int m0 = blockIdx.x * 128;
    const int n0 = blockIdx.y * 128;

    // staging: tile is 128 rows x 64B; 512 chunks of 16B; 2 per thread per matrix
    const int row_s[2] = { (tid) >> 2, (tid + 256) >> 2 };
    const int off_s[2] = { (tid & 3) * 16, (tid & 3) * 16 };

    const uint64_t Abase = (uint64_t)(size_t)A;
    const uint64_t Bbase = (uint64_t)(size_t)Bt;

    FragF acc[4][2];
    for (int mt = 0; mt < 4; ++mt)
        for (int nt = 0; nt < 2; ++nt)
            for (int i = 0; i < 8; ++i) acc[mt][nt].f[i] = 0.f;

#define ISSUE(buf, k0)                                                              \
    for (int i = 0; i < 2; ++i) {                                                   \
        int row = row_s[i], ob = off_s[i];                                          \
        async_b128(lds_off(&Abuf[buf][row][0]) + ob,                                \
                   Abase + (((size_t)(m0 + row) * K + (k0)) * 2 + ob));             \
        async_b128(lds_off(&Bbuf[buf][row][0]) + ob,                                \
                   Bbase + (((size_t)(n0 + row) * K + (k0)) * 2 + ob));             \
    }

    ISSUE(0, 0)
    wait_async0();
    __syncthreads();

    int it = 0;
    for (int k0 = 0; k0 < K; k0 += 32, ++it) {
        const int cur = it & 1;
        if (k0 + 32 < K) { ISSUE(cur ^ 1, k0 + 32) }

        FragB af[4], bf[2];
        for (int mt = 0; mt < 4; ++mt) {
            int r = wm * 64 + mt * 16 + lrow;
            af[mt].q[0] = *(const uint4*)&Abuf[cur][r][khalf];
            af[mt].q[1] = *(const uint4*)&Abuf[cur][r][khalf + 16];
        }
        for (int nt = 0; nt < 2; ++nt) {
            int nr = wn * 32 + nt * 16 + lrow;
            bf[nt].q[0] = *(const uint4*)&Bbuf[cur][nr][khalf];
            bf[nt].q[1] = *(const uint4*)&Bbuf[cur][nr][khalf + 16];
        }
        for (int mt = 0; mt < 4; ++mt)
            for (int nt = 0; nt < 2; ++nt)
                acc[mt][nt].v = wmma_bf16(af[mt], bf[nt], acc[mt][nt]);

        wait_async0();
        __syncthreads();
    }
#undef ISSUE

    for (int mt = 0; mt < 4; ++mt)
        for (int nt = 0; nt < 2; ++nt) {
            int col = n0 + wn * 32 + nt * 16 + lrow;
            for (int r = 0; r < 8; ++r)
                C[(size_t)(m0 + wm * 64 + mt * 16 + r + loff) * N + col] = acc[mt][nt].f[r];
        }
}

// ---------------------------------------------------------------- RMSNorm + 3-axis RoPE
__global__ __launch_bounds__(256) void rmsnorm_rope_kernel(float* __restrict__ qk,
                                                           unsigned short* __restrict__ out16,
                                                           const float* __restrict__ g,
                                                           const float* __restrict__ f_cs, const float* __restrict__ f_sn,
                                                           const float* __restrict__ h_cs, const float* __restrict__ h_sn,
                                                           const float* __restrict__ w_cs, const float* __restrict__ w_sn) {
    int t = blockIdx.x;
    float* row = qk + (size_t)t * DIM_;
    __shared__ float red[256];
    float s = 0.f;
    for (int i = threadIdx.x; i < DIM_; i += 256) { float v = row[i]; s += v * v; }
    red[threadIdx.x] = s;
    __syncthreads();
    for (int off = 128; off > 0; off >>= 1) {
        if (threadIdx.x < off) red[threadIdx.x] += red[threadIdx.x + off];
        __syncthreads();
    }
    float rs = rsqrtf(red[0] / DIM_ + 1e-6f);
    int fi = t >> 10, hi = (t >> 5) & 31, wi = t & 31;
    for (int p = threadIdx.x; p < DIM_ / 2; p += 256) {
        int d0 = 2 * p;
        int dd = d0 & (HD_ - 1);
        float c, sn;
        if (dd < 44)      { int j = dd >> 1;        c = f_cs[fi * 22 + j]; sn = f_sn[fi * 22 + j]; }
        else if (dd < 86) { int j = (dd - 44) >> 1; c = h_cs[hi * 21 + j]; sn = h_sn[hi * 21 + j]; }
        else              { int j = (dd - 86) >> 1; c = w_cs[wi * 21 + j]; sn = w_sn[wi * 21 + j]; }
        float e = row[d0]     * rs * g[d0];
        float o = row[d0 + 1] * rs * g[d0 + 1];
        float e2 = e * c - o * sn;
        float o2 = o * c + e * sn;
        row[d0] = e2; row[d0 + 1] = o2;
        out16[(size_t)t * DIM_ + d0]     = f32_to_bf16(e2);
        out16[(size_t)t * DIM_ + d0 + 1] = f32_to_bf16(o2);
    }
}

// ---------------------------------------------------------------- block-mean pooling
__global__ __launch_bounds__(256) void pool_kernel(const float* __restrict__ q,
                                                   float* __restrict__ pooled) {
    int nb = blockIdx.x;
    for (int c = threadIdx.x; c < DIM_; c += 256) {
        float s = 0.f;
        for (int t = 0; t < POOL_; ++t) s += q[(size_t)(nb * POOL_ + t) * DIM_ + c];
        pooled[nb * DIM_ + c] = s * (1.f / POOL_);
    }
}

// ---------------------------------------------------------------- draft block mask
__device__ __forceinline__ bool tiled_mask(int l, int m) {
    int p = l & 7, p2 = m & 7;
    int ra = p >> 1, ca = p & 1;
    int r2 = p2 >> 1, c2 = p2 & 1;
    return (r2 >= ra - 3) && (r2 <= ra + 2) && (c2 >= ca - 3) && (c2 <= ca + 2);
}

__global__ __launch_bounds__(256) void mask_kernel(const float* __restrict__ pq,
                                                   const float* __restrict__ pk,
                                                   int* __restrict__ bm) {
    __shared__ float sc[NH_][16][16];
    __shared__ float at[NH_][16][16];
    __shared__ float thr[NH_ * 2];
    int tid = threadIdx.x;
    for (int idx = tid; idx < NH_ * 256; idx += 256) {
        int h = idx >> 8, l = (idx >> 4) & 15, m = idx & 15;
        const float* a = pq + l * DIM_ + h * HD_;
        const float* b = pk + m * DIM_ + h * HD_;
        float s = 0.f;
        for (int d = 0; d < HD_; ++d) s += a[d] * b[d];
        sc[h][l][m] = s * 0.0883883476f;
    }
    __syncthreads();
    for (int idx = tid; idx < NH_ * 16; idx += 256) {
        int h = idx >> 4, l = idx & 15;
        float vals[16], mx = -1e30f;
        for (int m = 0; m < 16; ++m) {
            float v = tiled_mask(l, m) ? sc[h][l][m] : -1e30f;
            vals[m] = v; mx = fmaxf(mx, v);
        }
        float ssum = 0.f;
        for (int m = 0; m < 16; ++m) { float e = __expf(vals[m] - mx); vals[m] = e; ssum += e; }
        for (int m = 0; m < 16; ++m) at[h][l][m] = vals[m] / ssum;
    }
    __syncthreads();
    if (tid < NH_ * 2) {
        int h = tid >> 1, fr = tid & 1;
        float v[128];
        for (int i = 0; i < 128; ++i) v[i] = at[h][fr * 8 + (i >> 4)][i & 15];
        for (int i = 0; i <= 16; ++i) {
            int mj = i;
            for (int j = i + 1; j < 128; ++j) if (v[j] > v[mj]) mj = j;
            float t2 = v[i]; v[i] = v[mj]; v[mj] = t2;
        }
        thr[tid] = v[16];
    }
    __syncthreads();
    for (int idx = tid; idx < NH_ * 256; idx += 256) {
        int h = idx >> 8, l = (idx >> 4) & 15, m = idx & 15;
        bm[idx] = (at[h][l][m] > thr[h * 2 + (l >> 3)]) ? 1 : 0;
    }
}

// ---------------------------------------------------------------- block-sparse flash attention
// grid = (S/64, NH); 8 waves (2 along M x 4 along N); V pre-transposed bf16 [DIM][S]
__global__ __launch_bounds__(256) void attn_kernel(const unsigned short* __restrict__ qb,
                                                   const unsigned short* __restrict__ kbm,
                                                   const unsigned short* __restrict__ vT,
                                                   const int* __restrict__ bm,
                                                   unsigned short* __restrict__ outb) {
    const int qt = blockIdx.x;
    const int h  = blockIdx.y;
    const int tid  = threadIdx.x;
    const int lane = tid & 31;
    const int wave = tid >> 5;
    const int wm = wave >> 2, wn = wave & 3;
    const int lrow = lane & 15;
    const int loff = (lane < 16) ? 0 : 8;
    const int khalf = (lane < 16) ? 0 : 8;

    __shared__ __align__(16) unsigned short Plds[QT][64];
    __shared__ float pmax[QT][4], psum[QT][4];
    __shared__ float m_row[QT], l_row[QT], a_row[QT];

    for (int i = tid; i < QT; i += 256) { m_row[i] = -1e30f; l_row[i] = 0.f; }

    FragB qf[2][4];
    for (int mt = 0; mt < 2; ++mt) {
        const unsigned short* qp =
            qb + (size_t)(qt * QT + wm * 32 + mt * 16 + lrow) * DIM_ + h * HD_;
        for (int ks = 0; ks < 4; ++ks) {
            int k0 = ks * 32 + khalf;
            qf[mt][ks].q[0] = *(const uint4*)&qp[k0];
            qf[mt][ks].q[1] = *(const uint4*)&qp[k0 + 16];
        }
    }

    FragF oacc[2][2];
    for (int mt = 0; mt < 2; ++mt)
        for (int nt = 0; nt < 2; ++nt)
            for (int i = 0; i < 8; ++i) oacc[mt][nt].f[i] = 0.f;

    const int* brow = bm + (h * 16 + (qt * QT) / POOL_) * 16;
    __syncthreads();

    for (int kb_i = 0; kb_i < 16; ++kb_i) {
        if (!brow[kb_i]) continue;
        for (int half = 0; half < 2; ++half) {
            const int key0 = kb_i * POOL_ + half * 64;

            // S = Q x K^T over 64-key chunk (wave owns 16 keys)
            FragF sacc[2];
            for (int mt = 0; mt < 2; ++mt)
                for (int i = 0; i < 8; ++i) sacc[mt].f[i] = 0.f;
            {
                const unsigned short* kp =
                    kbm + (size_t)(key0 + wn * 16 + lrow) * DIM_ + h * HD_;
                for (int ks = 0; ks < 4; ++ks) {
                    FragB bf_;
                    int k0 = ks * 32 + khalf;
                    bf_.q[0] = *(const uint4*)&kp[k0];
                    bf_.q[1] = *(const uint4*)&kp[k0 + 16];
                    for (int mt = 0; mt < 2; ++mt)
                        sacc[mt].v = wmma_bf16(qf[mt][ks], bf_, sacc[mt]);
                }
            }
            for (int mt = 0; mt < 2; ++mt)
                for (int r = 0; r < 8; ++r) sacc[mt].f[r] *= 0.0883883476f;

            // partial row-max across the 16-lane N group
            float rmx[2][8];
            for (int mt = 0; mt < 2; ++mt)
                for (int r = 0; r < 8; ++r) {
                    float v = sacc[mt].f[r];
                    for (int x = 1; x < 16; x <<= 1) v = fmaxf(v, __shfl_xor(v, x, 32));
                    rmx[mt][r] = v;
                }
            if (lrow == 0)
                for (int mt = 0; mt < 2; ++mt)
                    for (int r = 0; r < 8; ++r)
                        pmax[wm * 32 + mt * 16 + r + loff][wn] = rmx[mt][r];
            __syncthreads();

            if (tid < QT) {
                float gm = fmaxf(fmaxf(pmax[tid][0], pmax[tid][1]),
                                 fmaxf(pmax[tid][2], pmax[tid][3]));
                float mo = m_row[tid];
                float mn = fmaxf(mo, gm);
                m_row[tid] = mn;
                a_row[tid] = __expf(mo - mn);
            }
            __syncthreads();

            // P = exp(S - m), write bf16 probs, partial row sums; rescale O
            float rsum[2][8];
            for (int mt = 0; mt < 2; ++mt)
                for (int r = 0; r < 8; ++r) {
                    int rloc = wm * 32 + mt * 16 + r + loff;
                    float p = __expf(sacc[mt].f[r] - m_row[rloc]);
                    Plds[rloc][wn * 16 + lrow] = f32_to_bf16(p);
                    float v = p;
                    for (int x = 1; x < 16; x <<= 1) v += __shfl_xor(v, x, 32);
                    rsum[mt][r] = v;
                }
            if (lrow == 0)
                for (int mt = 0; mt < 2; ++mt)
                    for (int r = 0; r < 8; ++r)
                        psum[wm * 32 + mt * 16 + r + loff][wn] = rsum[mt][r];
            for (int mt = 0; mt < 2; ++mt)
                for (int r = 0; r < 8; ++r) {
                    float al = a_row[wm * 32 + mt * 16 + r + loff];
                    oacc[mt][0].f[r] *= al;
                    oacc[mt][1].f[r] *= al;
                }
            __syncthreads();

            if (tid < QT)
                l_row[tid] = l_row[tid] * a_row[tid] +
                             psum[tid][0] + psum[tid][1] + psum[tid][2] + psum[tid][3];

            // O += P x V : A-frags from Plds, B-frags straight from V^T (global)
            for (int ks2 = 0; ks2 < 2; ++ks2) {
                FragB af[2], bf2[2];
                int kk0 = ks2 * 32 + khalf;
                for (int mt = 0; mt < 2; ++mt) {
                    int rr = wm * 32 + mt * 16 + lrow;
                    af[mt].q[0] = *(const uint4*)&Plds[rr][kk0];
                    af[mt].q[1] = *(const uint4*)&Plds[rr][kk0 + 16];
                }
                for (int nt = 0; nt < 2; ++nt) {
                    const unsigned short* vp =
                        vT + (size_t)(h * HD_ + wn * 32 + nt * 16 + lrow) * S_ + key0;
                    bf2[nt].q[0] = *(const uint4*)&vp[kk0];
                    bf2[nt].q[1] = *(const uint4*)&vp[kk0 + 16];
                }
                for (int mt = 0; mt < 2; ++mt)
                    for (int nt = 0; nt < 2; ++nt)
                        oacc[mt][nt].v = wmma_bf16(af[mt], bf2[nt], oacc[mt][nt]);
            }
            __syncthreads();
        }
    }

    for (int mt = 0; mt < 2; ++mt)
        for (int nt = 0; nt < 2; ++nt)
            for (int r = 0; r < 8; ++r) {
                int rloc = wm * 32 + mt * 16 + r + loff;
                float l = l_row[rloc];
                float val = (l > 0.f) ? oacc[mt][nt].f[r] / l : 0.f;
                int col = h * HD_ + wn * 32 + nt * 16 + lrow;
                outb[(size_t)(qt * QT + rloc) * DIM_ + col] = f32_to_bf16(val);
            }
}

// ---------------------------------------------------------------- launch
extern "C" void kernel_launch(void* const* d_in, const int* in_sizes, int n_in,
                              void* d_out, int out_size, void* d_ws, size_t ws_size,
                              hipStream_t stream) {
    const float* x    = (const float*)d_in[0];
    const float* wq   = (const float*)d_in[1];
    const float* wk   = (const float*)d_in[2];
    const float* wv   = (const float*)d_in[3];
    const float* wo   = (const float*)d_in[4];
    const float* gq   = (const float*)d_in[5];
    const float* gk   = (const float*)d_in[6];
    const float* f_cs = (const float*)d_in[7];
    const float* f_sn = (const float*)d_in[8];
    const float* h_cs = (const float*)d_in[9];
    const float* h_sn = (const float*)d_in[10];
    const float* w_cs = (const float*)d_in[11];
    const float* w_sn = (const float*)d_in[12];
    float* out = (float*)d_out;

    char* ws = (char*)d_ws;
    auto alloc = [&](size_t bytes) {
        char* p = ws;
        ws += (bytes + 255) & ~(size_t)255;
        return p;
    };
    const size_t NX = (size_t)S_ * DIM_;
    const size_t NW = (size_t)DIM_ * DIM_;

    unsigned short* xb   = (unsigned short*)alloc(NX * 2);
    unsigned short* wqT  = (unsigned short*)alloc(NW * 2);
    unsigned short* wkT  = (unsigned short*)alloc(NW * 2);
    unsigned short* wvT  = (unsigned short*)alloc(NW * 2);
    unsigned short* woT  = (unsigned short*)alloc(NW * 2);
    float*          qf   = (float*)alloc(NX * 4);
    float*          kf   = (float*)alloc(NX * 4);
    float*          vf   = (float*)alloc(NX * 4);
    unsigned short* q16  = (unsigned short*)alloc(NX * 2);
    unsigned short* k16  = (unsigned short*)alloc(NX * 2);
    unsigned short* vT16 = (unsigned short*)alloc(NX * 2);
    unsigned short* a16  = (unsigned short*)alloc(NX * 2);
    float*          pq   = (float*)alloc((size_t)16 * DIM_ * 4);
    float*          pk   = (float*)alloc((size_t)16 * DIM_ * 4);
    int*            bmv  = (int*)alloc((size_t)NH_ * 16 * 16 * 4);

    // x -> bf16 (row-major); weights -> bf16 transposed [N][K]
    cvt_bf16_kernel<<<(int)((NX + 255) / 256), 256, 0, stream>>>(x, xb, (int)NX);
    dim3 gt(DIM_ / 32, DIM_ / 32);
    transpose_cvt_kernel<<<gt, 256, 0, stream>>>(wq, wqT, DIM_, DIM_);
    transpose_cvt_kernel<<<gt, 256, 0, stream>>>(wk, wkT, DIM_, DIM_);
    transpose_cvt_kernel<<<gt, 256, 0, stream>>>(wv, wvT, DIM_, DIM_);
    transpose_cvt_kernel<<<gt, 256, 0, stream>>>(wo, woT, DIM_, DIM_);

    // Q/K/V projections (async double-buffered WMMA GEMM)
    dim3 gg(S_ / 128, DIM_ / 128);
    gemm_bf16_wmma<<<gg, 256, 0, stream>>>(xb, wqT, qf, S_, DIM_, DIM_);
    gemm_bf16_wmma<<<gg, 256, 0, stream>>>(xb, wkT, kf, S_, DIM_, DIM_);
    gemm_bf16_wmma<<<gg, 256, 0, stream>>>(xb, wvT, vf, S_, DIM_, DIM_);

    // RMSNorm + RoPE (in-place f32, emit bf16)
    rmsnorm_rope_kernel<<<S_, 256, 0, stream>>>(qf, q16, gq, f_cs, f_sn, h_cs, h_sn, w_cs, w_sn);
    rmsnorm_rope_kernel<<<S_, 256, 0, stream>>>(kf, k16, gk, f_cs, f_sn, h_cs, h_sn, w_cs, w_sn);

    // V -> bf16 transposed [DIM][S]
    transpose_cvt_kernel<<<dim3(DIM_ / 32, S_ / 32), 256, 0, stream>>>(vf, vT16, S_, DIM_);

    // draft block mask
    pool_kernel<<<16, 256, 0, stream>>>(qf, pq);
    pool_kernel<<<16, 256, 0, stream>>>(kf, pk);
    mask_kernel<<<1, 256, 0, stream>>>(pq, pk, bmv);

    // block-sparse flash attention
    attn_kernel<<<dim3(S_ / QT, NH_), 256, 0, stream>>>(q16, k16, vT16, bmv, a16);

    // output projection
    gemm_bf16_wmma<<<gg, 256, 0, stream>>>(a16, woT, out, S_, DIM_, DIM_);
}
